// VMEncoder_36945308680677
// MI455X (gfx1250) — matrix operand order repaired and checked by reading, not compiled, not run on hardware
//
#include <hip/hip_runtime.h>

typedef __attribute__((ext_vector_type(16))) _Float16 v16h;
typedef __attribute__((ext_vector_type(8)))  float    v8f;

#define RANK 12
#define RES 192
#define PLANE_STRIDE (RES * RES)   // elements per rank slice
#define OUTD 32
#define KREAL 36
#define KPAD 64
#define AROW 72                    // padded A row stride in f16 (144 B, 16B-aligned)
#define WAVE_CHUNK 4608            // bytes per wave: 32 A-rows * 144B; reused for 32x32 f32 D tile (4096B)

__device__ __forceinline__ unsigned int pk2(_Float16 a, _Float16 b) {
    union { _Float16 h[2]; unsigned int u; } x;
    x.h[0] = a; x.h[1] = b;
    return x.u;
}

__device__ __forceinline__ v16h lds_load16(const _Float16* p) {
    union { uint4 q[2]; v16h v; } u;
    u.q[0] = *(const uint4*)(p);
    u.q[1] = *(const uint4*)(p + 8);
    return u.v;
}

__global__ __launch_bounds__(256) void vm_encoder_kernel(
    const float* __restrict__ xyz,
    const float* __restrict__ pxy,
    const float* __restrict__ pxz,
    const float* __restrict__ pyz,
    const float* __restrict__ w,   // [32][36] row-major
    float* __restrict__ out,       // [N][32]
    int npts)
{
    __shared__ __align__(16) unsigned char ldsBuf[8 * WAVE_CHUNK]; // per-wave A tile, reused as D tile
    __shared__ __align__(16) _Float16 ldsW[OUTD * KPAD];           // wT[n][k] zero-padded to K=64

    const int tid  = threadIdx.x;
    const int lane = tid & 31;
    const int wave = tid >> 5;
    const int blockBase = blockIdx.x * 256;

    _Float16* Awave = (_Float16*)(ldsBuf + wave * WAVE_CHUNK);
    float*    Dwave = (float*)(ldsBuf + wave * WAVE_CHUNK);

    // ---- stage weights: ldsW[n][k] = w_mlp[n][k], zero-padded in K ----
    for (int idx = tid; idx < OUTD * KPAD; idx += 256) {
        int n = idx >> 6, k = idx & 63;
        float wv = (k < KREAL) ? w[n * KREAL + k] : 0.0f;
        ldsW[idx] = (_Float16)wv;
    }

    // ---- per-lane feature computation (one point per lane) ----
    int p  = blockBase + wave * 32 + lane;
    int pc = (p < npts) ? p : (npts - 1);
    const float* q = xyz + 3 * pc;
    float px = q[0], py = q[1], pz = q[2];

    _Float16 fr[KPAD];
    #pragma unroll
    for (int k = KREAL; k < KPAD; ++k) fr[k] = (_Float16)0.0f;

    const float* planes[3] = { pxy, pxz, pyz };
    const float  us[3]     = { px, px, py };
    const float  vsv[3]    = { py, pz, pz };

    #pragma unroll
    for (int pl = 0; pl < 3; ++pl) {
        const float* plane = planes[pl];
        float x = fminf(fmaxf(us[pl]  * 191.0f, 0.0f), 191.0f);
        float y = fminf(fmaxf(vsv[pl] * 191.0f, 0.0f), 191.0f);
        float xf = floorf(x), yf = floorf(y);
        int x0 = (int)xf, y0 = (int)yf;
        int x1 = (x0 < RES - 1) ? x0 + 1 : RES - 1;
        int y1 = (y0 < RES - 1) ? y0 + 1 : RES - 1;
        float wx = x - xf, wy = y - yf;
        float w00 = (1.0f - wx) * (1.0f - wy);
        float w01 = wx * (1.0f - wy);
        float w10 = (1.0f - wx) * wy;
        float w11 = wx * wy;
        const float* b00 = plane + (y0 * RES + x0);
        const float* b01 = plane + (y0 * RES + x1);
        const float* b10 = plane + (y1 * RES + x0);
        const float* b11 = plane + (y1 * RES + x1);
        #pragma unroll
        for (int r = 0; r < RANK; ++r) {
            int o = r * PLANE_STRIDE;   // immediate ioffset per rank (max 1.6 MB < 2^23)
            float s = b00[o] * w00 + b01[o] * w01 + b10[o] * w10 + b11[o] * w11;
            fr[pl * RANK + r] = (_Float16)s;
        }
    }

    // ---- store row to LDS in WMMA A-fragment order ----
    // dest groups of 8: [K0-7][K8-15][K16-23][K24-31][K32-39][K40-47][K48-55][K56-63]
    // so fragment (half, kstep) reads 16 contiguous f16 at offset kstep*32 + half*16.
    {
        _Float16* rowp = Awave + lane * AROW;
        constexpr int gmap[8] = {0, 1, 2, 3, 4, 5, 6, 7};
        #pragma unroll
        for (int d = 0; d < 8; ++d) {
            const int g = gmap[d];
            uint4 qq;
            qq.x = pk2(fr[g * 8 + 0], fr[g * 8 + 1]);
            qq.y = pk2(fr[g * 8 + 2], fr[g * 8 + 3]);
            qq.z = pk2(fr[g * 8 + 4], fr[g * 8 + 5]);
            qq.w = pk2(fr[g * 8 + 6], fr[g * 8 + 7]);
            *(uint4*)(rowp + d * 8) = qq;
        }
    }

    __syncthreads();   // for ldsW staging (A is wave-private)

    const int half = lane >> 4;
    const int nn   = lane & 15;

    // ---- load ALL fragments up front (A region is then reused for D) ----
    v16h b00v = lds_load16(&ldsW[(0 * 16 + nn) * KPAD +  0 + half * 16]);
    v16h b01v = lds_load16(&ldsW[(0 * 16 + nn) * KPAD + 32 + half * 16]);
    v16h b10v = lds_load16(&ldsW[(1 * 16 + nn) * KPAD +  0 + half * 16]);
    v16h b11v = lds_load16(&ldsW[(1 * 16 + nn) * KPAD + 32 + half * 16]);

    v16h a00 = lds_load16(Awave + (0 * 16 + nn) * AROW +  0 + half * 16);
    v16h a01 = lds_load16(Awave + (0 * 16 + nn) * AROW + 32 + half * 16);
    v16h a10 = lds_load16(Awave + (1 * 16 + nn) * AROW +  0 + half * 16);
    v16h a11 = lds_load16(Awave + (1 * 16 + nn) * AROW + 32 + half * 16);

    // ---- WMMA: 2 M-tiles x 2 N-tiles, K=64 in two chained steps; park D in LDS ----
    #pragma unroll
    for (int t = 0; t < 2; ++t) {
        v16h ak0 = (t == 0) ? a00 : a10;
        v16h ak1 = (t == 0) ? a01 : a11;
        #pragma unroll
        for (int nt = 0; nt < 2; ++nt) {
            v8f c = {};
            c = __builtin_amdgcn_wmma_f32_16x16x32_f16(
                    false, ak0, false, (nt == 0) ? b00v : b10v, (short)0, c, false, false);
            c = __builtin_amdgcn_wmma_f32_16x16x32_f16(
                    false, ak1, false, (nt == 0) ? b01v : b11v, (short)0, c, false, false);

            float* dst = Dwave + (t * 16 + half * 8) * OUTD + nt * 16 + nn;
            #pragma unroll
            for (int vv = 0; vv < 8; ++vv) dst[vv * OUTD] = c[vv];
        }
    }

    // ---- coalesced writeback: each pass stores 4 full rows (512B contiguous) ----
    const int subrow = lane >> 3;       // 0..3
    const int chunk  = lane & 7;        // 0..7 -> 16B column chunk
    const bool full  = (blockBase + 256) <= npts;   // uniform per block

    if (full) {
        #pragma unroll
        for (int j = 0; j < 8; ++j) {
            int lp = j * 4 + subrow;
            float4 d = *(const float4*)(Dwave + lp * OUTD + chunk * 4);
            size_t row = (size_t)(blockBase + wave * 32 + lp);
            *(float4*)(out + row * OUTD + chunk * 4) = d;
        }
    } else {
        #pragma unroll
        for (int j = 0; j < 8; ++j) {
            int lp = j * 4 + subrow;
            int row = blockBase + wave * 32 + lp;
            if (row < npts) {
                float4 d = *(const float4*)(Dwave + lp * OUTD + chunk * 4);
                *(float4*)(out + (size_t)row * OUTD + chunk * 4) = d;
            }
        }
    }
}

extern "C" void kernel_launch(void* const* d_in, const int* in_sizes, int n_in,
                              void* d_out, int out_size, void* d_ws, size_t ws_size,
                              hipStream_t stream) {
    const float* xyz = (const float*)d_in[0];
    const float* pxy = (const float*)d_in[1];
    const float* pxz = (const float*)d_in[2];
    const float* pyz = (const float*)d_in[3];
    const float* w   = (const float*)d_in[4];
    float* out = (float*)d_out;

    int npts = in_sizes[0] / 3;
    int nBlocks = (npts + 255) / 256;
    vm_encoder_kernel<<<nBlocks, 256, 0, stream>>>(xyz, pxy, pxz, pyz, w, out, npts);
}
